// MyDecoder_35897336660441
// MI455X (gfx1250) — matrix operand order repaired and checked by reading, not compiled, not run on hardware
//
#include <hip/hip_runtime.h>

typedef __attribute__((ext_vector_type(16))) __bf16       v16bf;
typedef __attribute__((ext_vector_type(8)))  __bf16       v8bf;
typedef __attribute__((ext_vector_type(8)))  float        v8f;
typedef __attribute__((ext_vector_type(4)))  unsigned int u32x4;
typedef __attribute__((ext_vector_type(8)))  int          i32x8;
typedef __attribute__((ext_vector_type(4)))  int          i32x4;

#define D_IN  256
#define D_HID 256
#define D_OUT 128

// ---------------- CSR build: histogram, scan, bucket ----------------
__global__ void k_zero_i32(int* __restrict__ p, int n) {
    int i = blockIdx.x * blockDim.x + threadIdx.x;
    if (i < n) p[i] = 0;
}
__global__ void k_hist(const int* __restrict__ dst, int* __restrict__ cnt, int nE) {
    int e = blockIdx.x * blockDim.x + threadIdx.x;
    if (e < nE) atomicAdd(&cnt[dst[e]], 1);
}
// dinv[i] = rsqrt(indeg + 1)   (self-loop adds 1)
__global__ void k_dinv(const int* __restrict__ cnt, float* __restrict__ dinv, int n) {
    int i = blockIdx.x * blockDim.x + threadIdx.x;
    if (i < n) dinv[i] = rsqrtf((float)cnt[i] + 1.0f);
}
// single block, 1024 threads: exclusive prefix sum -> rowptr[n+1], cursor[n]
__global__ __launch_bounds__(1024) void k_scan(const int* __restrict__ cnt,
                                               int* __restrict__ rowptr,
                                               int* __restrict__ cursor, int n) {
    __shared__ int part[1024];
    int tid = threadIdx.x;
    int per = (n + 1023) >> 10;
    int base = tid * per;
    int s = 0;
    for (int i = 0; i < per; ++i) { int j = base + i; if (j < n) s += cnt[j]; }
    part[tid] = s;
    __syncthreads();
    for (int off = 1; off < 1024; off <<= 1) {       // Hillis-Steele inclusive scan
        int v = (tid >= off) ? part[tid - off] : 0;
        __syncthreads();
        part[tid] += v;
        __syncthreads();
    }
    int run = (tid == 0) ? 0 : part[tid - 1];
    for (int i = 0; i < per; ++i) {
        int j = base + i;
        if (j < n) { rowptr[j] = run; cursor[j] = run; run += cnt[j]; }
    }
    if (tid == 0) rowptr[n] = part[1023];
}
__global__ void k_bucket(const int* __restrict__ src, const int* __restrict__ dst,
                         int* __restrict__ cursor, int* __restrict__ col, int nE) {
    int e = blockIdx.x * blockDim.x + threadIdx.x;
    if (e < nE) {
        int p = atomicAdd(&cursor[dst[e]], 1);
        col[p] = src[e];
    }
}

// ---------------- weight convert: W[K,N] f32 -> Wt[N,K] bf16 ----------------
__global__ void k_wt_bf16(const float* __restrict__ W, __bf16* __restrict__ Wt, int K, int N) {
    int idx = blockIdx.x * blockDim.x + threadIdx.x;
    if (idx < K * N) {
        int k = idx / N;
        int n = idx - k * N;
        Wt[(size_t)n * K + k] = (__bf16)W[idx];
    }
}

// ---------------- WMMA GEMM: C[M,N] = A[M,K](f32) * Wt[N,K](bf16)^T ----------------
// Block = 256 threads = 8 waves; wave w computes rows [m0, m0+16), cols [n0, n0+64).
// B slab (64 cols x K bf16) staged once per block into LDS by the Tensor Data Mover,
// with TDM padding (16B per 512B row -> 528B stride) for conflict-free ds_load_b128.
// Fragment layouts per CDNA5 ISA 7.12.2 (wave32, 16x16x32 bf16, f32 accum).
template <int NT>
__global__ __launch_bounds__(256) void k_gemm_wmma(
    const float* __restrict__ A, const __bf16* __restrict__ Wt,
    float* __restrict__ C, int M, int N, int K)
{
    constexpr int ROW_BYTES = 2 * 256;          // K == 256 bf16 per row
    constexpr int LDS_STRIDE = ROW_BYTES + 16;  // TDM pad: 4-bank skew per row
    __shared__ char shB[NT * 16 * LDS_STRIDE];  // 64 rows -> 33,792 B

    const int wave = threadIdx.x >> 5;
    const int lane = threadIdx.x & 31;
    const int lrow = lane & 15;
    const int lhi  = lane >> 4;                 // 0: lanes 0-15, 1: lanes 16-31
    const int n0   = blockIdx.y * (NT * 16);

    // --- TDM: DMA Wt rows [n0, n0+NT*16), all K, into LDS (one wave issues) ---
    if (wave == 0) {
        uint64_t ga = (uint64_t)(const void*)(Wt + (size_t)n0 * K);
        u32x4 g0 = { 1u,                                  // count=1, load, no gather
                     0u,                                  // lds_addr = 0 (shB is sole LDS)
                     (unsigned)(ga & 0xffffffffu),
                     (unsigned)(((ga >> 32) & 0x1ffffffu) | (2u << 30)) }; // type=2
        i32x8 g1 = { (int)0x07910000,   // data_size=2B | pad_en | pad_intv=6(512B) | pad_amt=3(16B)
                     (int)(K << 16),    // tensor_dim0 = K
                     (int)(N << 16),    // tensor_dim1 = N (rows of Wt)
                     (int)(K << 16),    // tile_dim0 = K
                     NT * 16,           // tile_dim1 = 64 rows, tile_dim2 = 0
                     K,                 // tensor_dim0_stride = K
                     0, 0 };
        i32x4 gz  = { 0, 0, 0, 0 };
#if defined(__clang_major__) && __clang_major__ >= 23
        i32x8 gz8 = { 0, 0, 0, 0, 0, 0, 0, 0 };
        __builtin_amdgcn_tensor_load_to_lds(g0, g1, gz, gz, gz8, 0);
#else
        __builtin_amdgcn_tensor_load_to_lds(g0, g1, gz, gz, 0);
#endif
        __builtin_amdgcn_s_wait_tensorcnt(0);
    }
    __syncthreads();

    const int m0 = blockIdx.x * 128 + wave * 16;
    if (m0 < M) {                               // wave-uniform: EXEC all-ones for WMMA
        v8f acc[NT] = {};
        const float* arow = A + (size_t)(m0 + lrow) * K;

        for (int k0 = 0; k0 < K; k0 += 32) {
            // A fragment: elems 0-7 -> K = k0 + 8*lhi + e ; elems 8-15 -> +16
            v8f f0 = *(const v8f*)(arow + k0 + lhi * 8);
            v8f f1 = *(const v8f*)(arow + k0 + 16 + lhi * 8);
            if (k0 + 32 < K) __builtin_prefetch(arow + k0 + 32, 0, 0);

            // Issue ALL B-fragment LDS loads first (one clause, staged waits)
            v16bf b[NT];
#pragma unroll
            for (int t = 0; t < NT; ++t) {
                const char* bp = shB + (size_t)(t * 16 + lrow) * LDS_STRIDE
                                     + lhi * 16 + k0 * 2;
                v8bf blo = *(const v8bf*)bp;
                v8bf bhi = *(const v8bf*)(bp + 32);
#pragma unroll
                for (int i = 0; i < 8; ++i) { b[t][i] = blo[i]; b[t][8 + i] = bhi[i]; }
            }
            v16bf a;
#pragma unroll
            for (int i = 0; i < 8; ++i) {
                a[i]     = (__bf16)f0[i];
                a[8 + i] = (__bf16)f1[i];
            }
            // Back-to-back WMMA issue on the matrix pipe
#pragma unroll
            for (int t = 0; t < NT; ++t) {
                acc[t] = __builtin_amdgcn_wmma_f32_16x16x32_bf16(
                    /*neg_a=*/false, a, /*neg_b=*/false, b[t],
                    /*c_mod=*/(short)0, acc[t], /*reuse_a=*/false, /*reuse_b=*/false);
            }
        }
        // C/D layout: VGPR r -> row m0 + 8*lhi + r, col n0 + t*16 + lrow
#pragma unroll
        for (int t = 0; t < NT; ++t) {
#pragma unroll
            for (int r = 0; r < 8; ++r) {
                C[(size_t)(m0 + lhi * 8 + r) * N + (n0 + t * 16 + lrow)] = acc[t][r];
            }
        }
    }
}

// ---------------- fused aggregation (gather form, no atomics) ----------------
// out[i,c] = relu?( dinv[i] * sum_{s in N(i)} dinv[s]*xw[s,c] + dinv[i]^2*xw[i,c] + bias[c] )
// one block per node, one thread per channel; in-edge rows are coalesced reads.
template <int RELU>
__global__ void k_gather(const float* __restrict__ xw, const int* __restrict__ rowptr,
                         const int* __restrict__ col, const float* __restrict__ dinv,
                         const float* __restrict__ bias, float* __restrict__ out, int d) {
    int node = blockIdx.x;
    int c    = threadIdx.x;
    int beg = rowptr[node], end = rowptr[node + 1];
    float di = dinv[node];
    float s = 0.0f;
    for (int p = beg; p < end; ++p) {
        int sn = col[p];                           // block-uniform -> scalar load
        s += dinv[sn] * xw[(size_t)sn * d + c];    // coalesced row gather (L2-resident)
    }
    float r = di * s + di * di * xw[(size_t)node * d + c] + bias[c];
    if (RELU) r = fmaxf(r, 0.0f);
    out[(size_t)node * d + c] = r;
}

// ---------------- launcher ----------------
extern "C" void kernel_launch(void* const* d_in, const int* in_sizes, int n_in,
                              void* d_out, int out_size, void* d_ws, size_t ws_size,
                              hipStream_t stream) {
    const float* z  = (const float*)d_in[0];
    const int*   ei = (const int*)  d_in[1];
    const float* W1 = (const float*)d_in[2];
    const float* b1 = (const float*)d_in[3];
    const float* W2 = (const float*)d_in[4];
    const float* b2 = (const float*)d_in[5];
    float* out = (float*)d_out;

    const int n  = in_sizes[0] / D_IN;    // 50000
    const int nE = in_sizes[1] / 2;       // 800000
    const int* src = ei;
    const int* dst = ei + nE;

    // workspace carve-out (256B aligned)
    char* ws = (char*)d_ws;
    size_t off = 0;
    auto carve = [&](size_t bytes) -> char* {
        char* p = ws + off;
        off = (off + bytes + 255) & ~(size_t)255;
        return p;
    };
    float*  dinv   = (float*) carve((size_t)n * 4);
    int*    cnt    = (int*)   carve((size_t)n * 4);
    int*    rowptr = (int*)   carve((size_t)(n + 1) * 4);
    int*    cursor = (int*)   carve((size_t)n * 4);
    int*    col    = (int*)   carve((size_t)nE * 4);
    __bf16* W1t    = (__bf16*)carve((size_t)D_IN  * D_HID * 2);
    __bf16* W2t    = (__bf16*)carve((size_t)D_HID * D_OUT * 2);
    float*  bufA   = (float*) carve((size_t)n * D_HID * 4);  // xw1, later xw2
    float*  bufB   = (float*) carve((size_t)n * D_HID * 4);  // h

    // ---- CSR by destination (built once, reused by both layers) ----
    k_zero_i32<<<(n  + 255) / 256, 256, 0, stream>>>(cnt, n);
    k_hist    <<<(nE + 255) / 256, 256, 0, stream>>>(dst, cnt, nE);
    k_dinv    <<<(n  + 255) / 256, 256, 0, stream>>>(cnt, dinv, n);
    k_scan    <<<1, 1024, 0, stream>>>(cnt, rowptr, cursor, n);
    k_bucket  <<<(nE + 255) / 256, 256, 0, stream>>>(src, dst, cursor, col, nE);

    // weights -> bf16, transposed to [N,K] for contiguous B fragments
    k_wt_bf16<<<(D_IN  * D_HID + 255) / 256, 256, 0, stream>>>(W1, W1t, D_IN,  D_HID);
    k_wt_bf16<<<(D_HID * D_OUT + 255) / 256, 256, 0, stream>>>(W2, W2t, D_HID, D_OUT);

    // layer 1: xw1 = z @ W1 (WMMA, TDM-staged B); gather-aggregate + bias + relu
    dim3 g1((n + 127) / 128, D_HID / 64);
    k_gemm_wmma<4><<<g1, 256, 0, stream>>>(z, W1t, bufA, n, D_HID, D_IN);
    k_gather<1><<<n, D_HID, 0, stream>>>(bufA, rowptr, col, dinv, b1, bufB, D_HID);

    // layer 2: xw2 = h @ W2 (reuse bufA); gather-aggregate + bias
    dim3 g2((n + 127) / 128, D_OUT / 64);
    k_gemm_wmma<4><<<g2, 256, 0, stream>>>(bufB, W2t, bufA, n, D_OUT, D_HID);
    k_gather<0><<<n, D_OUT, 0, stream>>>(bufA, rowptr, col, dinv, b2, out, D_OUT);
}
